// PeriodicSetTransformer_24704651887015
// MI455X (gfx1250) — matrix operand order repaired and checked by reading, not compiled, not run on hardware
//
#include <hip/hip_runtime.h>
#include <hip/hip_bf16.h>
#include <math.h>

// ---------------------------------------------------------------------------
// PeriodicSetTransformer for MI455X (gfx1250, wave32, WMMA bf16 16x16x32)
// ---------------------------------------------------------------------------

#define B_   32
#define N_   512
#define F_   101
#define S_   10
#define E_   128
#define H_   4
#define EH_  512
#define HD_  128
#define L_   3

typedef __attribute__((ext_vector_type(16))) __bf16 v16bf;
typedef __attribute__((ext_vector_type(8)))  float  v8f;
typedef __attribute__((ext_vector_type(2)))  __bf16 bf16x2;

#define GF_BIAS     1
#define GF_ACC      2
#define GF_SOFTPLUS 4

#define BM 64
#define BN 128
#define KT 32
#define LDSW 18   // LDS row stride in dwords (36 bf16): padded, conflict-free

// Native bf16 converts: let clang select the gfx1250 VALU cvt (RNE).
__device__ __forceinline__ unsigned short bf_bits(float f) {
  return __builtin_bit_cast(unsigned short, (__bf16)f);
}
__device__ __forceinline__ unsigned pack_bf2(float lo, float hi) {
  bf16x2 p;
  p.x = (__bf16)lo;
  p.y = (__bf16)hi;
  return __builtin_bit_cast(unsigned, p);
}

__device__ __forceinline__ void mma_step(const unsigned* __restrict__ ldsA32,
                                         const unsigned* __restrict__ ldsB32,
                                         int waveM, int waveN, int lm, int half,
                                         v8f acc[2][2]) {
  v16bf afrag[2], bfrag[2];
#pragma unroll
  for (int tM = 0; tM < 2; ++tM) {
    int row = waveM * 32 + tM * 16 + lm;
#pragma unroll
    for (int vv = 0; vv < 8; ++vv) {
      // A 16x32 bf16 layout: k = (vv>>2)*16 + half*8 + (vv&3)*2 -> dword index
      int kd = ((vv >> 2) << 3) + (half << 2) + (vv & 3);
      bf16x2 pr = __builtin_bit_cast(bf16x2, ldsA32[row * LDSW + kd]);
      afrag[tM][2 * vv]     = pr.x;
      afrag[tM][2 * vv + 1] = pr.y;
    }
  }
#pragma unroll
  for (int tN = 0; tN < 2; ++tN) {
    int col = waveN * 32 + tN * 16 + lm;
#pragma unroll
    for (int vv = 0; vv < 8; ++vv) {
      // B 32x16 bf16 layout: k = (vv>>2)*16 + half*8 + (vv&3)*2 -> dword index
      int kd = ((vv >> 2) << 3) + (half << 2) + (vv & 3);
      bf16x2 pr = __builtin_bit_cast(bf16x2, ldsB32[col * LDSW + kd]);
      bfrag[tN][2 * vv]     = pr.x;
      bfrag[tN][2 * vv + 1] = pr.y;
    }
  }
#pragma unroll
  for (int tM = 0; tM < 2; ++tM)
#pragma unroll
    for (int tN = 0; tN < 2; ++tN)
      acc[tM][tN] = __builtin_amdgcn_wmma_f32_16x16x32_bf16(
          false, afrag[tM], false, bfrag[tN], (short)0, acc[tM][tN], false, false);
}

// ---------------------------------------------------------------------------
// Batched GEMM: C[m,n] = sum_k A[m,k]*B[k,n]  (+bias, +Cin, softplus)
// Contract (guaranteed by all launches below):
//   M % 64 == 0, N % 128 == 0, sAk == 1, sAm even,
//   B is either k-contiguous (sBk==1, sBn even) or n-contiguous (sBn==1).
// ---------------------------------------------------------------------------
__global__ __launch_bounds__(256)
void gemm_bf16_wmma(const float* __restrict__ A, const float* __restrict__ Bm,
                    const float* __restrict__ bias, const float* Cin, float* C,
                    int M, int N, int K,
                    long sAm, long batA,
                    long sBk, long sBn, long batB,
                    long ldc, long batC, int flags)
{
  __shared__ __align__(16) unsigned ldsA32[BM * LDSW];
  __shared__ __align__(16) unsigned ldsB32[BN * LDSW];

  const int t     = threadIdx.x;
  const int wave  = t >> 5;
  const int lane  = t & 31;
  const int waveM = wave >> 2;   // 0..1
  const int waveN = wave & 3;    // 0..3
  const int lm    = lane & 15;
  const int half  = lane >> 4;

  const long bz = blockIdx.z;
  A  += bz * batA;
  Bm += bz * batB;
  C  += bz * batC;
  if (Cin) Cin += bz * batC;

  const int blockM = blockIdx.y * BM;
  const int blockN = blockIdx.x * BN;

  v8f acc[2][2] = {};

  const int kfull = K / KT;
  const int krem  = K - kfull * KT;
  const bool bKcontig = (sBk == 1);

  // per-thread staging coordinates (hoisted)
  const int ar = t >> 4, akp = t & 15;         // A: pair p = i*256+t -> row p>>4, kpair p&15
  const float* Arow0 = A + (long)(blockM + ar) * sAm;              // rows ar, ar+16, ar+32, ar+48
  for (int kt = 0; kt < kfull; ++kt) {
    const int k0 = kt * KT;
    // ---- stage A (64x32): 4 float2 loads -> 4 packed ds_store_b32 ----
#pragma unroll
    for (int i = 0; i < 4; ++i) {
      const float2 v = *(const float2*)(Arow0 + (long)i * 16 * sAm + (k0 + 2 * akp));
      ldsA32[(ar + i * 16) * LDSW + akp] = pack_bf2(v.x, v.y);
    }
    if (kt + 1 < kfull)
      __builtin_prefetch(Arow0 + (k0 + KT + 2 * akp), 0, 1);

    // ---- stage B (32x128 -> LDS [n][k]) ----
    if (bKcontig) {
      // element (k,n) at k + n*sBn : float2 along k
      const int bn = t >> 4, bkp = t & 15;
#pragma unroll
      for (int i = 0; i < 8; ++i) {
        int n = bn + i * 16;
        const float2 v = *(const float2*)(Bm + (long)(blockN + n) * sBn + (k0 + 2 * bkp));
        ldsB32[n * LDSW + bkp] = pack_bf2(v.x, v.y);
      }
      if (kt + 1 < kfull)
        __builtin_prefetch(Bm + (long)(blockN + bn) * sBn + (k0 + KT + 2 * bkp), 0, 1);
    } else {
      // element (k,n) at k*sBk + n : two coalesced b32 loads along n
      const int bn = t & 127, bkp0 = t >> 7;   // p = kp*128 + n
#pragma unroll
      for (int i = 0; i < 8; ++i) {
        int kp = bkp0 + i * 2;
        const float* s0 = Bm + (long)(k0 + 2 * kp) * sBk + (blockN + bn);
        ldsB32[bn * LDSW + kp] = pack_bf2(s0[0], s0[sBk]);
      }
      if (kt + 1 < kfull)
        __builtin_prefetch(Bm + (long)(k0 + KT + 2 * bkp0) * sBk + (blockN + bn), 0, 1);
    }
    __syncthreads();
    mma_step(ldsA32, ldsB32, waveM, waveN, lm, half, acc);
    __syncthreads();
  }

  // ---- cold remainder k-tile (zero-padded, guarded) ----
  if (krem) {
    const int k0 = kfull * KT;
    unsigned short* ldsA16 = (unsigned short*)ldsA32;
    unsigned short* ldsB16 = (unsigned short*)ldsB32;
#pragma unroll
    for (int i = 0; i < 8; ++i) {
      int e = i * 256 + t;
      int r = e >> 5, kk = e & 31;
      int gk = k0 + kk;
      float v = (kk < krem) ? A[(long)(blockM + r) * sAm + gk] : 0.f;
      ldsA16[(r * LDSW) * 2 + kk] = bf_bits(v);
    }
#pragma unroll
    for (int i = 0; i < 16; ++i) {
      int e = i * 256 + t;
      int n = e >> 5, kk = e & 31;
      int gk = k0 + kk;
      float v = (kk < krem) ? Bm[(long)gk * sBk + (long)(blockN + n) * sBn] : 0.f;
      ldsB16[(n * LDSW) * 2 + kk] = bf_bits(v);
    }
    __syncthreads();
    mma_step(ldsA32, ldsB32, waveM, waveN, lm, half, acc);
    __syncthreads();
  }

  // ---- epilogue (no M/N guards: exact tiling by contract) ----
#pragma unroll
  for (int tM = 0; tM < 2; ++tM) {
#pragma unroll
    for (int tN = 0; tN < 2; ++tN) {
      int gn = blockN + waveN * 32 + tN * 16 + lm;
#pragma unroll
      for (int r = 0; r < 8; ++r) {
        int gm = blockM + waveM * 32 + tM * 16 + r + (half << 3);
        float v = acc[tM][tN][r];
        if (flags & GF_BIAS) v += bias[gn];
        if (flags & GF_ACC)  v += Cin[(long)gm * ldc + gn];
        if (flags & GF_SOFTPLUS) v = (v > 20.f) ? v : log1pf(__expf(v));
        C[(long)gm * ldc + gn] = v;
      }
    }
  }
}

// ---------------------------------------------------------------------------
// Elementwise / reduction kernels
// ---------------------------------------------------------------------------
__global__ void build_de_kernel(const float* __restrict__ str_fea,
                                float* __restrict__ de, long total) {
  long idx = (long)blockIdx.x * 256 + threadIdx.x;
  if (idx >= total) return;
  long fs = idx % ((F_ - 1) * S_);
  long bn = idx / ((F_ - 1) * S_);
  int f = (int)(fs / S_), s = (int)(fs % S_);
  float d = str_fea[bn * F_ + 1 + f];
  float v = 1.f - (d - (float)s * (1.f / (float)S_));
  de[idx] = v * v;
}

__global__ void gather_comp_kernel(const int* __restrict__ comp_fea,
                                   const float* __restrict__ af_table,
                                   float* __restrict__ compg, long total) {
  long idx = (long)blockIdx.x * 256 + threadIdx.x;
  if (idx >= total) return;
  int c = (int)(idx % 92);
  long bn = idx / 92;
  compg[idx] = af_table[(long)comp_fea[bn] * 92 + c];
}

// LayerNorm over last dim 128; one block (128 thr) per row; optional add input
__global__ __launch_bounds__(128)
void layernorm128_kernel(const float* __restrict__ in, const float* add,
                         const float* __restrict__ g, const float* __restrict__ b,
                         float* __restrict__ out) {
  long row = blockIdx.x;
  int t = threadIdx.x;
  float x = in[row * E_ + t];
  if (add) x += add[row * E_ + t];
  float s = x, s2 = x * x;
#pragma unroll
  for (int o = 16; o > 0; o >>= 1) {
    s  += __shfl_xor(s, o, 32);
    s2 += __shfl_xor(s2, o, 32);
  }
  __shared__ float ps[4], ps2[4];
  int w = t >> 5;
  if ((t & 31) == 0) { ps[w] = s; ps2[w] = s2; }
  __syncthreads();
  float ts  = ps[0] + ps[1] + ps[2] + ps[3];
  float ts2 = ps2[0] + ps2[1] + ps2[2] + ps2[3];
  float mean = ts * (1.f / E_);
  float var  = ts2 * (1.f / E_) - mean * mean;
  float inv  = rsqrtf(var + 1e-5f);
  out[row * E_ + t] = (x - mean) * inv * g[t] + b[t];
}

__device__ __forceinline__ float blk_reduce(float v, float* sred, bool is_max) {
#pragma unroll
  for (int o = 16; o > 0; o >>= 1) {
    float u = __shfl_xor(v, o, 32);
    v = is_max ? fmaxf(v, u) : (v + u);
  }
  int w = threadIdx.x >> 5;
  if ((threadIdx.x & 31) == 0) sred[w] = v;
  __syncthreads();
  float r = sred[0];
#pragma unroll
  for (int i = 1; i < 8; ++i) r = is_max ? fmaxf(r, sred[i]) : (r + sred[i]);
  __syncthreads();
  return r;
}

// per-(b,i) row: aw += softmax(scale*logits_h masked)/H  (h==0 initializes)
__global__ __launch_bounds__(256)
void softmax_accum_kernel(const float* __restrict__ logits,
                          const float* __restrict__ str_fea,
                          float* __restrict__ aw, int h) {
  __shared__ float sred[8];
  long bi = blockIdx.x;
  long b = bi >> 9, i = bi & (N_ - 1);
  int t = threadIdx.x;
  const float scale = 0.08838834764831845f; // 1/sqrt(128)
  int j0 = t, j1 = t + 256;
  const float* lrow = logits + (b * N_ + i) * N_;
  float w0 = str_fea[(b * N_ + j0) * F_];
  float w1 = str_fea[(b * N_ + j1) * F_];
  float l0 = (w0 == 0.f) ? -1e9f : lrow[j0] * scale;
  float l1 = (w1 == 0.f) ? -1e9f : lrow[j1] * scale;
  float m = blk_reduce(fmaxf(l0, l1), sred, true);
  float e0 = __expf(l0 - m), e1 = __expf(l1 - m);
  float s = blk_reduce(e0 + e1, sred, false);
  float inv = 0.25f / s;   // mean over H folded in
  float* arow = aw + (b * N_ + i) * N_;
  if (h == 0) { arow[j0] = e0 * inv;  arow[j1] = e1 * inv; }
  else        { arow[j0] += e0 * inv; arow[j1] += e1 * inv; }
}

// aw[i,j] = aw[i,j]*w_row[j] / sum_j(aw[i,j]*w_row[j])
__global__ __launch_bounds__(256)
void aw_normalize_kernel(float* __restrict__ aw, const float* __restrict__ str_fea) {
  __shared__ float sred[8];
  long bi = blockIdx.x;
  long b = bi >> 9, i = bi & (N_ - 1);
  int t = threadIdx.x;
  int j0 = t, j1 = t + 256;
  float* arow = aw + (b * N_ + i) * N_;
  float v0 = arow[j0] * str_fea[(b * N_ + j0) * F_];
  float v1 = arow[j1] * str_fea[(b * N_ + j1) * F_];
  float s = blk_reduce(v0 + v1, sred, false);
  float inv = 1.f / s;
  arow[j0] = v0 * inv;
  arow[j1] = v1 * inv;
}

// pooled[b,e] = sum_n w_row[b,n] * (x[b,n,e] + x_init[b,n,e])
__global__ __launch_bounds__(128)
void pool_weighted_kernel(const float* __restrict__ x, const float* __restrict__ xi,
                          const float* __restrict__ str_fea, float* __restrict__ pooled) {
  long b = blockIdx.x;
  int e = threadIdx.x;
  float acc = 0.f;
  for (int n = 0; n < N_; ++n) {
    float w = str_fea[(b * N_ + n) * F_];
    long o = (b * N_ + n) * E_ + e;
    acc += w * (x[o] + xi[o]);
  }
  pooled[b * E_ + e] = acc;
}

__global__ __launch_bounds__(128)
void final_proj_kernel(const float* __restrict__ pln, const float* __restrict__ fw,
                       const float* __restrict__ fb, float* __restrict__ out) {
  long b = blockIdx.x;
  int t = threadIdx.x;
  float v = pln[b * E_ + t] * fw[t];
#pragma unroll
  for (int o = 16; o > 0; o >>= 1) v += __shfl_xor(v, o, 32);
  __shared__ float ps[4];
  if ((t & 31) == 0) ps[t >> 5] = v;
  __syncthreads();
  if (t == 0) out[b] = ps[0] + ps[1] + ps[2] + ps[3] + fb[0];
}

// ---------------------------------------------------------------------------
// Host orchestration
// ---------------------------------------------------------------------------
static inline void gemm(hipStream_t s, const float* A, const float* Bm,
                        const float* bias, const float* Cin, float* C,
                        int M, int N, int K,
                        long sAm, long batA,
                        long sBk, long sBn, long batB,
                        long ldc, long batC, int nbat, int flags) {
  dim3 g((N + BN - 1) / BN, (M + BM - 1) / BM, nbat);
  gemm_bf16_wmma<<<g, 256, 0, s>>>(A, Bm, bias, Cin, C, M, N, K,
                                   sAm, batA, sBk, sBn, batB, ldc, batC, flags);
}

extern "C" void kernel_launch(void* const* d_in, const int* in_sizes, int n_in,
                              void* d_out, int out_size, void* d_ws, size_t ws_size,
                              hipStream_t stream) {
  const float* str_fea  = (const float*)d_in[0];
  const int*   comp_fea = (const int*)  d_in[1];
  // d_in[2] cell_fea unused by reference
  const float* af_table = (const float*)d_in[3];
  const float* comp_w   = (const float*)d_in[4];
  const float* comp_b   = (const float*)d_in[5];
  const float* emb_w    = (const float*)d_in[6];
  const float* emb_b    = (const float*)d_in[7];
  const float* ln_g     = (const float*)d_in[8];
  const float* ln_b     = (const float*)d_in[9];
  const float* wq       = (const float*)d_in[10];
  const float* bq       = (const float*)d_in[11];
  const float* wk       = (const float*)d_in[12];
  const float* bk       = (const float*)d_in[13];
  const float* wv       = (const float*)d_in[14];
  const float* bv       = (const float*)d_in[15];
  const float* inproj_w = (const float*)d_in[16];
  const float* inproj_b = (const float*)d_in[17];
  const float* out_w    = (const float*)d_in[18];
  const float* out_b    = (const float*)d_in[19];
  const float* ffn_w    = (const float*)d_in[20];
  const float* ffn_b    = (const float*)d_in[21];
  const float* ln2_g    = (const float*)d_in[22];
  const float* ln2_b    = (const float*)d_in[23];
  const float* final_w  = (const float*)d_in[24];
  const float* final_b  = (const float*)d_in[25];

  const long BN_TOK = (long)B_ * N_;          // 16384
  const long SZ_X   = BN_TOK * E_;            // 2.10M floats
  const long SZ_BIG = BN_TOK * EH_;           // 8.39M floats

  float* W = (float*)d_ws;
  float* x      = W;                 // +SZ_X
  float* x_init = x      + SZ_X;
  float* xn     = x_init + SZ_X;
  float* o1     = xn     + SZ_X;
  float* o2     = o1     + SZ_X;
  float* aw     = o2     + SZ_X;     // +SZ_BIG
  float* att    = aw     + SZ_BIG;
  float* qin    = att    + SZ_BIG;
  float* kin    = qin    + SZ_BIG;
  float* vin    = kin    + SZ_BIG;
  float* qb     = vin    + SZ_BIG;
  float* kb     = qb     + SZ_BIG;
  float* lgt    = kb     + SZ_BIG;
  float* pooled = lgt    + SZ_BIG;   // B_*E_
  float* pooled_ln = pooled + (long)B_ * E_;
  // embedding scratch aliased onto qin/kin (de: 16.38M < 16.78M) and vin (compg)
  float* de    = qin;
  float* compg = vin;

  // ---------- embedding ----------
  long tot_de = BN_TOK * (long)((F_ - 1) * S_);
  build_de_kernel<<<(unsigned)((tot_de + 255) / 256), 256, 0, stream>>>(str_fea, de, tot_de);
  long tot_cg = BN_TOK * 92L;
  gather_comp_kernel<<<(unsigned)((tot_cg + 255) / 256), 256, 0, stream>>>(comp_fea, af_table, compg, tot_cg);

  // x = compg @ comp_w + comp_b
  gemm(stream, compg, comp_w, comp_b, nullptr, x, (int)BN_TOK, E_, 92,
       92, 0,   E_, 1, 0,   E_, 0, 1, GF_BIAS);
  // x += de @ emb_w + emb_b
  gemm(stream, de, emb_w, emb_b, x, x, (int)BN_TOK, E_, (F_ - 1) * S_,
       (F_ - 1) * S_, 0,   E_, 1, 0,   E_, 0, 1, GF_BIAS | GF_ACC);
  hipMemcpyAsync(x_init, x, SZ_X * sizeof(float), hipMemcpyDeviceToDevice, stream);

  // ---------- transformer layers ----------
  for (int l = 0; l < L_; ++l) {
    const float* lg = ln_g + (long)l * E_;
    const float* lb = ln_b + (long)l * E_;
    layernorm128_kernel<<<(unsigned)BN_TOK, 128, 0, stream>>>(x, nullptr, lg, lb, xn);

    const float* wql = wq + (long)l * E_ * EH_;
    const float* wkl = wk + (long)l * E_ * EH_;
    const float* wvl = wv + (long)l * E_ * EH_;
    gemm(stream, xn, wql, bq + (long)l * EH_, nullptr, qin, (int)BN_TOK, EH_, E_,
         E_, 0,   EH_, 1, 0,   EH_, 0, 1, GF_BIAS);
    gemm(stream, xn, wkl, bk + (long)l * EH_, nullptr, kin, (int)BN_TOK, EH_, E_,
         E_, 0,   EH_, 1, 0,   EH_, 0, 1, GF_BIAS);
    gemm(stream, xn, wvl, bv + (long)l * EH_, nullptr, vin, (int)BN_TOK, EH_, E_,
         E_, 0,   EH_, 1, 0,   EH_, 0, 1, GF_BIAS);

    const float* ipw = inproj_w + (long)l * EH_ * (3 * EH_);
    const float* ipb = inproj_b + (long)l * (3 * EH_);
    gemm(stream, qin, ipw,        ipb,        nullptr, qb, (int)BN_TOK, EH_, EH_,
         EH_, 0,   3 * EH_, 1, 0,   EH_, 0, 1, GF_BIAS);
    gemm(stream, kin, ipw + EH_,  ipb + EH_,  nullptr, kb, (int)BN_TOK, EH_, EH_,
         EH_, 0,   3 * EH_, 1, 0,   EH_, 0, 1, GF_BIAS);

    // per-head: logits_h = Qh @ Kh^T ; accumulate mean-softmax into aw
    for (int h = 0; h < H_; ++h) {
      gemm(stream, qb + (long)h * HD_, kb + (long)h * HD_, nullptr, nullptr, lgt,
           N_, N_, HD_,
           EH_, (long)N_ * EH_,         // A: q[b, i, h*HD + k]
           1, EH_, (long)N_ * EH_,      // B: k[b, j, h*HD + k] -> (k,n=j), k-contiguous
           N_, (long)N_ * N_, B_, 0);
      softmax_accum_kernel<<<(unsigned)(B_ * N_), 256, 0, stream>>>(lgt, str_fea, aw, h);
    }
    aw_normalize_kernel<<<(unsigned)(B_ * N_), 256, 0, stream>>>(aw, str_fea);

    // att = aw @ vin
    gemm(stream, aw, vin, nullptr, nullptr, att, N_, EH_, N_,
         N_, (long)N_ * N_,
         EH_, 1, (long)N_ * EH_,
         EH_, (long)N_ * EH_, B_, 0);

    // o1 = x + att @ out_w + out_b
    hipMemcpyAsync(o1, x, SZ_X * sizeof(float), hipMemcpyDeviceToDevice, stream);
    gemm(stream, att, out_w + (long)l * EH_ * E_, out_b + (long)l * E_, o1, o1,
         (int)BN_TOK, E_, EH_,
         EH_, 0,   E_, 1, 0,   E_, 0, 1, GF_BIAS | GF_ACC);

    // o2 = softplus(LN(o1) @ ffn_w + ffn_b)
    layernorm128_kernel<<<(unsigned)BN_TOK, 128, 0, stream>>>(o1, nullptr, lg, lb, xn);
    gemm(stream, xn, ffn_w + (long)l * E_ * E_, ffn_b + (long)l * E_, nullptr, o2,
         (int)BN_TOK, E_, E_,
         E_, 0,   E_, 1, 0,   E_, 0, 1, GF_BIAS | GF_SOFTPLUS);

    // x = LN(o1 + o2)
    layernorm128_kernel<<<(unsigned)BN_TOK, 128, 0, stream>>>(o1, o2, lg, lb, x);
  }

  // ---------- pooling + head ----------
  pool_weighted_kernel<<<B_, 128, 0, stream>>>(x, x_init, str_fea, pooled);
  layernorm128_kernel<<<B_, 128, 0, stream>>>(pooled, nullptr, ln2_g, ln2_b, pooled_ln);
  final_proj_kernel<<<B_, 128, 0, stream>>>(pooled_ln, final_w, final_b, (float*)d_out);
}
